// NodeImportance_41678362640781
// MI455X (gfx1250) — matrix operand order repaired and checked by reading, not compiled, not run on hardware
//
#include <hip/hip_runtime.h>
#include <hip/hip_bf16.h>

typedef __attribute__((ext_vector_type(16))) _Float16 v16h;
typedef __attribute__((ext_vector_type(8)))  float    v8f;

#define WAVE 32

// ---------------------------------------------------------------------------
// Kernel 0: zero the scalar aggregation buffer s[N]
// ---------------------------------------------------------------------------
__global__ void zero_f32(float* p, int n) {
    int i = blockIdx.x * blockDim.x + threadIdx.x;
    if (i < n) p[i] = 0.0f;
}

// ---------------------------------------------------------------------------
// Kernel 1: per-node dot products via WMMA.
//   p[i] = dot(x[i], w_rel),  q[i] = dot(x[i], w_root)
// One wave handles 16 nodes. A = 16x32 f16 tile of x (per 32-feature chunk),
// B = 32x16 f16 with column0 = w_rel chunk, column1 = w_root chunk, rest 0.
// Accumulate 4 chunks (D=128) in f32, D-matrix column0/1 -> p/q.
// ---------------------------------------------------------------------------
__global__ void __launch_bounds__(256)
wmma_node_dots(const float* __restrict__ x,
               const float* __restrict__ wrel,
               const float* __restrict__ wroot,
               float* __restrict__ p, float* __restrict__ q, int N) {
    const int lane = threadIdx.x & (WAVE - 1);
    const int wid  = blockIdx.x * (blockDim.x / WAVE) + (threadIdx.x / WAVE);
    const int base = wid * 16;
    if (base >= N) return;                      // wave-uniform guard (EXEC stays full)

    const int m  = lane & 15;                   // A-row / B-col index
    const int hi = lane >> 4;                   // lane half select

    int node = base + m;
    if (node > N - 1) node = N - 1;             // clamp loads for a partial tail group
    const float* xr = x + (size_t)node * 128;

    v8f acc;
#pragma unroll
    for (int r = 0; r < 8; ++r) acc[r] = 0.0f;

#pragma unroll
    for (int c = 0; c < 4; ++c) {
        // ---- A tile: lane needs K = {8*hi..8*hi+7} and {16+8*hi..16+8*hi+7}
        const float4* r0 = (const float4*)(xr + 32 * c + 8 * hi);
        const float4* r1 = (const float4*)(xr + 32 * c + 16 + 8 * hi);
        float4 f0 = r0[0], f1 = r0[1];
        float4 f2 = r1[0], f3 = r1[1];
        v16h a;
        a[0]=(_Float16)f0.x; a[1]=(_Float16)f0.y; a[2]=(_Float16)f0.z; a[3]=(_Float16)f0.w;
        a[4]=(_Float16)f1.x; a[5]=(_Float16)f1.y; a[6]=(_Float16)f1.z; a[7]=(_Float16)f1.w;
        a[8]=(_Float16)f2.x; a[9]=(_Float16)f2.y; a[10]=(_Float16)f2.z; a[11]=(_Float16)f2.w;
        a[12]=(_Float16)f3.x; a[13]=(_Float16)f3.y; a[14]=(_Float16)f3.z; a[15]=(_Float16)f3.w;

        // ---- B tile: col m, K = {8*hi..8*hi+7} and {16+8*hi..16+8*hi+7} (same as A)
        v16h b;
#pragma unroll
        for (int e = 0; e < 16; ++e) b[e] = (_Float16)0.0f;
        if (m < 2) {
            const float* wv = (m == 0) ? wrel : wroot;
            const float4* w4 = (const float4*)(wv + 32 * c + 8 * hi);
            float4 g0 = w4[0], g1 = w4[1], g2 = w4[4], g3 = w4[5];
            b[0]=(_Float16)g0.x; b[1]=(_Float16)g0.y; b[2]=(_Float16)g0.z; b[3]=(_Float16)g0.w;
            b[4]=(_Float16)g1.x; b[5]=(_Float16)g1.y; b[6]=(_Float16)g1.z; b[7]=(_Float16)g1.w;
            b[8]=(_Float16)g2.x; b[9]=(_Float16)g2.y; b[10]=(_Float16)g2.z; b[11]=(_Float16)g2.w;
            b[12]=(_Float16)g3.x; b[13]=(_Float16)g3.y; b[14]=(_Float16)g3.z; b[15]=(_Float16)g3.w;
        }

        acc = __builtin_amdgcn_wmma_f32_16x16x32_f16(
            /*neg_a=*/false, a, /*neg_b=*/false, b,
            /*c_mod=*/(short)0, acc, /*reuse_a=*/false, /*reuse_b=*/false);
    }

    // D(m_row, n): lane = n + 16*(m_row>=8), VGPR r = m_row & 7.
    // Column 0 (lanes 0/16) -> p, column 1 (lanes 1/17) -> q.
    if (m < 2) {
        float* dst = (m == 0 ? p : q) + base + 8 * hi;
#pragma unroll
        for (int r = 0; r < 8; ++r) {
            if (base + 8 * hi + r < N) dst[r] = acc[r];
        }
    }
}

// ---------------------------------------------------------------------------
// Kernel 2: edge scatter of the scalar relation dot: s[dst] += p[src]
// ---------------------------------------------------------------------------
__global__ void edge_scatter(const int* __restrict__ ei,
                             const float* __restrict__ p,
                             float* __restrict__ s, int E) {
    int e = blockIdx.x * blockDim.x + threadIdx.x;
    if (e < E) {
        int src = ei[e];
        int dst = ei[E + e];
        atomicAdd(&s[dst], p[src]);
    }
}

// ---------------------------------------------------------------------------
// Kernel 3: fuse scores, build descending-sort keys (ties -> lower index)
// ---------------------------------------------------------------------------
__global__ void score_and_keys(const float* __restrict__ s,
                               const float* __restrict__ q,
                               const float* __restrict__ closeness,
                               const float* __restrict__ degree,
                               const float* __restrict__ b_gnn,
                               const float* __restrict__ w_close,
                               const float* __restrict__ w_deg,
                               const float* __restrict__ w_score,
                               const float* __restrict__ bias,
                               float* __restrict__ score,
                               unsigned long long* __restrict__ keys,
                               int N, int M) {
    int i = blockIdx.x * blockDim.x + threadIdx.x;
    if (i >= M) return;
    if (i < N) {
        float g = fmaxf(s[i] + q[i] + b_gnn[0], 0.0f);
        float cent = closeness[i] * w_close[0] + degree[i] * w_deg[0] + bias[0];
        float sc = fmaxf(g * w_score[0] + cent, 0.0f);
        score[i] = sc;
        unsigned u = __float_as_uint(sc);
        u = (u & 0x80000000u) ? ~u : (u | 0x80000000u);   // order-preserving map
        keys[i] = ((unsigned long long)u << 32) | (unsigned)(0xFFFFFFFFu - (unsigned)i);
    } else {
        keys[i] = 0ull;                                   // pad: smaller than any real key
    }
}

// ---------------------------------------------------------------------------
// Kernel 4: one bitonic compare-exchange stage (descending sort)
// ---------------------------------------------------------------------------
__global__ void bitonic_stage(unsigned long long* __restrict__ keys, int j, int k) {
    int i = blockIdx.x * blockDim.x + threadIdx.x;
    int ixj = i ^ j;
    if (ixj > i) {
        unsigned long long a = keys[i];
        unsigned long long b = keys[ixj];
        bool up = (i & k) == 0;
        if (up ? (a < b) : (a > b)) { keys[i] = b; keys[ixj] = a; }
    }
}

// ---------------------------------------------------------------------------
// Kernel 5: gather + scale output rows, emit perm. One wave per output row.
// ---------------------------------------------------------------------------
__global__ void __launch_bounds__(256)
gather_out(const float* __restrict__ x,
           const float* __restrict__ score,
           const unsigned long long* __restrict__ keys,
           float* __restrict__ out, int K, int out_size) {
    int row  = blockIdx.x * (blockDim.x / WAVE) + (threadIdx.x / WAVE);
    int lane = threadIdx.x & (WAVE - 1);
    if (row >= K) return;
    unsigned idx = 0xFFFFFFFFu - (unsigned)(keys[row] & 0xFFFFFFFFull);
    float sc = score[idx];
    const float4* xr = (const float4*)(x + (size_t)idx * 128);
    float4 v = xr[lane];
    float4* orow = (float4*)(out + (size_t)row * 128);
    orow[lane] = make_float4(v.x * sc, v.y * sc, v.z * sc, v.w * sc);
    if (lane == 0) {
        size_t poff = (size_t)K * 128 + row;             // perm (as float)
        if (poff < (size_t)out_size) out[poff] = (float)idx;
    }
}

// ---------------------------------------------------------------------------
// Kernel 6: zero the batch_out tail of d_out
// ---------------------------------------------------------------------------
__global__ void zero_tail(float* out, int start, int total) {
    int i = blockIdx.x * blockDim.x + threadIdx.x;
    if (start + i < total) out[start + i] = 0.0f;
}

// ---------------------------------------------------------------------------
extern "C" void kernel_launch(void* const* d_in, const int* in_sizes, int n_in,
                              void* d_out, int out_size, void* d_ws, size_t ws_size,
                              hipStream_t stream) {
    const float* x         = (const float*)d_in[0];
    const int*   ei        = (const int*)d_in[1];
    const float* closeness = (const float*)d_in[2];
    const float* degree    = (const float*)d_in[3];
    const float* wrel      = (const float*)d_in[4];
    const float* wroot     = (const float*)d_in[5];
    const float* b_gnn     = (const float*)d_in[6];
    const float* w_close   = (const float*)d_in[7];
    const float* w_deg     = (const float*)d_in[8];
    const float* w_score   = (const float*)d_in[9];
    const float* bias      = (const float*)d_in[10];
    float* out = (float*)d_out;

    const int N = in_sizes[2];            // 50000 nodes
    const int E = in_sizes[1] / 2;        // 800000 edges
    const int K = (N + 1) / 2;            // 25000
    int M = 1; while (M < N) M <<= 1;     // 65536 (padded sort size)

    // workspace layout (bytes)
    char* ws = (char*)d_ws;
    float* p_buf  = (float*)(ws + 0);
    float* q_buf  = (float*)(ws + (size_t)N * 4);
    float* s_buf  = (float*)(ws + (size_t)N * 8);
    float* sc_buf = (float*)(ws + (size_t)N * 12);
    unsigned long long* keys = (unsigned long long*)(ws + (size_t)N * 16);

    // 0) zero aggregation buffer
    zero_f32<<<(N + 255) / 256, 256, 0, stream>>>(s_buf, N);

    // 1) WMMA node dots: p = x.w_rel, q = x.w_root
    {
        int waves = (N + 15) / 16;
        int blocks = (waves + 7) / 8;     // 8 waves / 256-thread block
        wmma_node_dots<<<blocks, 256, 0, stream>>>(x, wrel, wroot, p_buf, q_buf, N);
    }

    // 2) scalar edge scatter-add
    edge_scatter<<<(E + 255) / 256, 256, 0, stream>>>(ei, p_buf, s_buf, E);

    // 3) score fusion + sort keys
    score_and_keys<<<(M + 255) / 256, 256, 0, stream>>>(
        s_buf, q_buf, closeness, degree, b_gnn, w_close, w_deg, w_score, bias,
        sc_buf, keys, N, M);

    // 4) bitonic sort (descending), 512 KB key array is L2-resident
    for (int k = 2; k <= M; k <<= 1)
        for (int j = k >> 1; j > 0; j >>= 1)
            bitonic_stage<<<M / 256, 256, 0, stream>>>(keys, j, k);

    // 5) gather + scale top-K rows, write perm
    gather_out<<<(K + 7) / 8, 256, 0, stream>>>(x, sc_buf, keys, out, K, out_size);

    // 6) zero batch_out tail
    {
        int start = K * 128 + K;
        int cnt = out_size - start;
        if (cnt > 0) zero_tail<<<(cnt + 255) / 256, 256, 0, stream>>>(out, start, out_size);
    }
}
